// MultiGAT_69106023793066
// MI455X (gfx1250) — matrix operand order repaired and checked by reading, not compiled, not run on hardware
//
#include <hip/hip_runtime.h>
#include <hip/hip_bf16.h>
#include <stdint.h>

typedef __attribute__((ext_vector_type(16))) _Float16 v16h;
typedef __attribute__((ext_vector_type(8)))  _Float16 v8h;
typedef __attribute__((ext_vector_type(8)))  float    v8f;

#define IN_DIM   128
#define HID_DIM  64
#define OUT_DIM  32
#define HEADS    8
#define NEG_SLOPE 0.2f

// ---------------------------------------------------------------------------
// Monotone float<->uint encoding so float max reduces to atomicMax(uint)
// ---------------------------------------------------------------------------
__device__ __forceinline__ unsigned fenc(float f) {
    unsigned u = __float_as_uint(f);
    return (u & 0x80000000u) ? ~u : (u | 0x80000000u);
}
__device__ __forceinline__ float fdec(unsigned u) {
    u = (u & 0x80000000u) ? (u & 0x7FFFFFFFu) : ~u;
    return __uint_as_float(u);
}
#define FENC_NEG_INF 0x007FFFFFu   // fenc(-inf)

// ---------------------------------------------------------------------------
// Weight pre-transpose + f32->f16 convert: Wt[c][k] = (f16) W[k][c]
// (tiny: <= 128x64; runs once per launch, keeps GEMM B-loads contiguous)
// ---------------------------------------------------------------------------
__global__ void wtrans_kernel(const float* __restrict__ W, _Float16* __restrict__ Wt,
                              int K, int NCOL)
{
    int i = blockIdx.x * blockDim.x + threadIdx.x;
    if (i >= K * NCOL) return;
    int k = i / NCOL, c = i - k * NCOL;
    Wt[(size_t)c * K + k] = (_Float16)W[i];
}

// ---------------------------------------------------------------------------
// Dense projection Y[n,NCOL] = X[n,K] @ W[K,NCOL] with v_wmma_f32_16x16x32_f16
// One wave per 16x16 output tile. A: f32 rows gathered as float4, converted.
// B: pre-transposed f16 rows, two b128 loads per fragment.
// ---------------------------------------------------------------------------
template <int K, int NCOL>
__global__ __launch_bounds__(256) void gemm_wmma_kernel(
    const float* __restrict__ X, const _Float16* __restrict__ Wt,
    float* __restrict__ Y, int nRows)
{
    const int lane = threadIdx.x & 31;
    const int wave = threadIdx.x >> 5;
    const int colTiles = NCOL / 16;
    const int rowTiles = (nRows + 15) >> 4;
    const int tile = blockIdx.x * 8 + wave;          // wave-uniform
    if (tile >= rowTiles * colTiles) return;         // whole wave exits together
    const int tr = tile / colTiles;
    const int tc = tile - tr * colTiles;
    const int r0 = tr << 4, c0 = tc << 4;

    const int m = lane & 15;   // A-row within tile / B,C,D column within tile
    const int g = lane >> 4;   // half-wave group

    int rr = r0 + m; if (rr >= nRows) rr = nRows - 1;      // clamped gather row
    const float*    __restrict__ xrow = X + (size_t)rr * K;
    const _Float16* __restrict__ wrow = Wt + (size_t)(c0 + m) * K;

    v8f acc = {};
    for (int k0 = 0; k0 < K; k0 += 32) {
        // ---- A fragment: halves 0..7 -> K = k0+g*8+i ; 8..15 -> K = k0+16+g*8+i
        const float4* xv = (const float4*)(xrow + k0);
        float4 p0 = xv[2 * g + 0];
        float4 p1 = xv[2 * g + 1];
        float4 p2 = xv[4 + 2 * g + 0];
        float4 p3 = xv[4 + 2 * g + 1];
        v16h a;
        a[0]  = (_Float16)p0.x; a[1]  = (_Float16)p0.y; a[2]  = (_Float16)p0.z; a[3]  = (_Float16)p0.w;
        a[4]  = (_Float16)p1.x; a[5]  = (_Float16)p1.y; a[6]  = (_Float16)p1.z; a[7]  = (_Float16)p1.w;
        a[8]  = (_Float16)p2.x; a[9]  = (_Float16)p2.y; a[10] = (_Float16)p2.z; a[11] = (_Float16)p2.w;
        a[12] = (_Float16)p3.x; a[13] = (_Float16)p3.y; a[14] = (_Float16)p3.z; a[15] = (_Float16)p3.w;
        // ---- B fragment: halves 0..7 -> K = k0+g*8+i ; 8..15 -> K = k0+16+g*8+i, column c0+m (pre-transposed)
        const v8h* wv = (const v8h*)(wrow + k0 + g * 8);
        v8h b0 = wv[0], b1 = wv[2];
        v16h b = __builtin_shufflevector(b0, b1, 0, 1, 2, 3, 4, 5, 6, 7,
                                                 8, 9, 10, 11, 12, 13, 14, 15);
        acc = __builtin_amdgcn_wmma_f32_16x16x32_f16(
            /*neg_a=*/false, a, /*neg_b=*/false, b,
            /*c_mod=*/(short)0, acc, /*reuse_a=*/false, /*reuse_b=*/false);
    }
    // C/D layout: VGPR i holds row (i + 8*g), column = lane&15
    if (r0 + 16 <= nRows) {      // fast path: full tile (always taken when 16|n)
#pragma unroll
        for (int i = 0; i < 8; ++i)
            Y[(size_t)(r0 + i + g * 8) * NCOL + c0 + m] = acc[i];
    } else {
#pragma unroll
        for (int i = 0; i < 8; ++i) {
            int row = r0 + i + g * 8;
            if (row < nRows) Y[(size_t)row * NCOL + c0 + m] = acc[i];
        }
    }
}

// ---------------------------------------------------------------------------
// Per-node attention halves: score(e,h) = as_dst[dst,h] + as_src[src,h]
// ---------------------------------------------------------------------------
__global__ void node_scores_kernel(const float* __restrict__ H,
                                   const float* __restrict__ att,
                                   float* __restrict__ as_src,
                                   float* __restrict__ as_dst,
                                   int n, int heads, int C)
{
    int i = blockIdx.x * blockDim.x + threadIdx.x;
    if (i >= n * heads) return;
    int node = i / heads, h = i - node * heads;
    const float* hp = H + (size_t)node * heads * C + (size_t)h * C;
    const float* ad = att + (size_t)h * 2 * C;   // dst part: att[0,h,:C]
    const float* as = ad + C;                    // src part: att[0,h,C:]
    float sd = 0.f, ss = 0.f;
    for (int c = 0; c < C; ++c) { float v = hp[c]; sd += v * ad[c]; ss += v * as[c]; }
    as_dst[i] = sd;
    as_src[i] = ss;
}

__global__ void fill_u32_kernel(unsigned* __restrict__ p, unsigned v, long long cnt) {
    long long i = (long long)blockIdx.x * blockDim.x + threadIdx.x;
    if (i < cnt) p[i] = v;
}
__global__ void fill_f32_kernel(float* __restrict__ p, float v, long long cnt) {
    long long i = (long long)blockIdx.x * blockDim.x + threadIdx.x;
    if (i < cnt) p[i] = v;
}

__device__ __forceinline__ bool edge_ends(const long long* __restrict__ srcI,
                                          const long long* __restrict__ dstI,
                                          int e, int E_, int& s, int& d)
{
    if (e < E_) {
        s = (int)srcI[e]; d = (int)dstI[e];
        return s != d;              // original self-loop -> masked (weight==0 exactly)
    }
    s = d = e - E_;                 // appended fresh self-loop
    return true;
}

__global__ void edge_max_kernel(const long long* __restrict__ srcI,
                                const long long* __restrict__ dstI,
                                int E_, int n,
                                const float* __restrict__ as_src,
                                const float* __restrict__ as_dst,
                                unsigned* __restrict__ amax, int heads)
{
    int e = blockIdx.x * blockDim.x + threadIdx.x;
    if (e >= E_ + n) return;
    int s, d;
    if (!edge_ends(srcI, dstI, e, E_, s, d)) return;
    for (int h = 0; h < heads; ++h) {
        float a = as_dst[d * heads + h] + as_src[s * heads + h];
        a = a > 0.f ? a : a * NEG_SLOPE;
        atomicMax(&amax[d * heads + h], fenc(a));
    }
}

__global__ void edge_denom_kernel(const long long* __restrict__ srcI,
                                  const long long* __restrict__ dstI,
                                  int E_, int n,
                                  const float* __restrict__ as_src,
                                  const float* __restrict__ as_dst,
                                  const unsigned* __restrict__ amax,
                                  float* __restrict__ denom, int heads)
{
    int e = blockIdx.x * blockDim.x + threadIdx.x;
    if (e >= E_ + n) return;
    int s, d;
    if (!edge_ends(srcI, dstI, e, E_, s, d)) return;
    for (int h = 0; h < heads; ++h) {
        float a = as_dst[d * heads + h] + as_src[s * heads + h];
        a = a > 0.f ? a : a * NEG_SLOPE;
        float ex = __expf(a - fdec(amax[d * heads + h]));
        atomicAdd(&denom[d * heads + h], ex);
    }
}

// One thread per (edge, 4-channel group): coalesced b128 gather of H[src],
// softmax weight computed once per group, 4 scalar f32 atomics (L2-resident).
__global__ void edge_agg_kernel(const long long* __restrict__ srcI,
                                const long long* __restrict__ dstI,
                                int E_, int n,
                                const float* __restrict__ as_src,
                                const float* __restrict__ as_dst,
                                const unsigned* __restrict__ amax,
                                const float* __restrict__ denom,
                                const float* __restrict__ H,
                                float* __restrict__ out,
                                int heads, int C)
{
    const int HC = heads * C;
    const int G  = HC >> 2;                          // float4 groups per edge
    long long gid = (long long)blockIdx.x * blockDim.x + threadIdx.x;
    long long total = (long long)(E_ + n) * G;
    if (gid >= total) return;
    int e = (int)(gid / G);
    int q = (int)(gid - (long long)e * G);
    int s, d;
    if (!edge_ends(srcI, dstI, e, E_, s, d)) return;
    int h = (q << 2) / C;
    float a = as_dst[d * heads + h] + as_src[s * heads + h];
    a = a > 0.f ? a : a * NEG_SLOPE;
    float w = __expf(a - fdec(amax[d * heads + h])) / denom[d * heads + h];
    float4 hv = ((const float4*)(H + (size_t)s * HC))[q];
    float* op = out + (size_t)d * HC + (q << 2);
    atomicAdd(op + 0, hv.x * w);
    atomicAdd(op + 1, hv.y * w);
    atomicAdd(op + 2, hv.z * w);
    atomicAdd(op + 3, hv.w * w);
}

__global__ void bias_act_kernel(float* __restrict__ Y, const float* __restrict__ b,
                                long long total, int dimMask, int doElu)
{
    long long i = (long long)blockIdx.x * blockDim.x + threadIdx.x;
    if (i >= total) return;
    float v = Y[i] + b[(int)i & dimMask];
    if (doElu) v = v > 0.f ? v : (__expf(v) - 1.f);
    Y[i] = v;
}

// ---------------------------------------------------------------------------
extern "C" void kernel_launch(void* const* d_in, const int* in_sizes, int n_in,
                              void* d_out, int out_size, void* d_ws, size_t ws_size,
                              hipStream_t stream)
{
    const float*     x    = (const float*)d_in[0];
    const long long* ei   = (const long long*)d_in[1];   // int64 [2,E]
    const float*     W1   = (const float*)d_in[2];
    const float*     att1 = (const float*)d_in[3];
    const float*     b1   = (const float*)d_in[4];
    const float*     W2   = (const float*)d_in[5];
    const float*     att2 = (const float*)d_in[6];
    const float*     b2   = (const float*)d_in[7];
    const float*     W3   = (const float*)d_in[8];
    const float*     att3 = (const float*)d_in[9];
    const float*     b3   = (const float*)d_in[10];

    const int n  = in_sizes[0] / IN_DIM;
    const int E_ = in_sizes[1] / 2;
    const long long* srcI = ei;
    const long long* dstI = ei + E_;

    // ---- workspace carve-up (~90 MB) ----
    char* ws = (char*)d_ws; size_t off = 0;
    auto carve = [&](size_t bytes) { void* p = ws + off; off += (bytes + 255) & ~(size_t)255; return p; };
    float*    hbuf   = (float*)carve((size_t)n * HID_DIM * sizeof(float));
    float*    featA  = (float*)carve((size_t)n * HID_DIM * sizeof(float));
    float*    featB  = (float*)carve((size_t)n * HID_DIM * sizeof(float));
    float*    as_src = (float*)carve((size_t)n * HEADS * sizeof(float));
    float*    as_dst = (float*)carve((size_t)n * HEADS * sizeof(float));
    unsigned* amax   = (unsigned*)carve((size_t)n * HEADS * sizeof(unsigned));
    float*    denom  = (float*)carve((size_t)n * HEADS * sizeof(float));
    _Float16* wt1    = (_Float16*)carve((size_t)IN_DIM  * HID_DIM * sizeof(_Float16));
    _Float16* wt2    = (_Float16*)carve((size_t)HID_DIM * HID_DIM * sizeof(_Float16));
    _Float16* wt3    = (_Float16*)carve((size_t)HID_DIM * OUT_DIM * sizeof(_Float16));
    float*    outF   = (float*)d_out;

    const int T = 256;
    const int Etot = E_ + n;
    auto blocks = [&](long long cnt) { return (unsigned)((cnt + T - 1) / T); };

    // weight transpose + f16 convert (tiny, once per launch)
    wtrans_kernel<<<blocks(IN_DIM * HID_DIM),  T, 0, stream>>>(W1, wt1, IN_DIM,  HID_DIM);
    wtrans_kernel<<<blocks(HID_DIM * HID_DIM), T, 0, stream>>>(W2, wt2, HID_DIM, HID_DIM);
    wtrans_kernel<<<blocks(HID_DIM * OUT_DIM), T, 0, stream>>>(W3, wt3, HID_DIM, OUT_DIM);

    auto run_layer = [&](const float* Xin, const _Float16* Wt, int which,
                         const float* att, const float* bias, float* Hproj,
                         float* aggOut, int heads, int C, int doElu) {
        const int HC = heads * C;
        int tiles = ((n + 15) / 16) * (HC / 16);
        if (which == 1)
            gemm_wmma_kernel<IN_DIM, HID_DIM><<<(tiles + 7) / 8, T, 0, stream>>>(Xin, Wt, Hproj, n);
        else if (which == 2)
            gemm_wmma_kernel<HID_DIM, HID_DIM><<<(tiles + 7) / 8, T, 0, stream>>>(Xin, Wt, Hproj, n);
        else
            gemm_wmma_kernel<HID_DIM, OUT_DIM><<<(tiles + 7) / 8, T, 0, stream>>>(Xin, Wt, Hproj, n);
        node_scores_kernel<<<blocks((long long)n * heads), T, 0, stream>>>(
            Hproj, att, as_src, as_dst, n, heads, C);
        fill_u32_kernel<<<blocks((long long)n * heads), T, 0, stream>>>(amax, FENC_NEG_INF, (long long)n * heads);
        fill_f32_kernel<<<blocks((long long)n * heads), T, 0, stream>>>(denom, 0.f, (long long)n * heads);
        fill_f32_kernel<<<blocks((long long)n * HC), T, 0, stream>>>(aggOut, 0.f, (long long)n * HC);
        edge_max_kernel<<<blocks(Etot), T, 0, stream>>>(srcI, dstI, E_, n, as_src, as_dst, amax, heads);
        edge_denom_kernel<<<blocks(Etot), T, 0, stream>>>(srcI, dstI, E_, n, as_src, as_dst, amax, denom, heads);
        edge_agg_kernel<<<blocks((long long)Etot * (HC >> 2)), T, 0, stream>>>(
            srcI, dstI, E_, n, as_src, as_dst, amax, denom, Hproj, aggOut, heads, C);
        bias_act_kernel<<<blocks((long long)n * HC), T, 0, stream>>>(
            aggOut, bias, (long long)n * HC, HC - 1, doElu);
    };

    // Layer 1: x[N,128] -> featA[N,64], 8 heads x 8 ch, ELU
    run_layer(x,     wt1, 1, att1, b1, hbuf, featA, HEADS, HID_DIM / HEADS, 1);
    // Layer 2: featA -> featB[N,64], 8 heads x 8 ch, ELU
    run_layer(featA, wt2, 2, att2, b2, hbuf, featB, HEADS, HID_DIM / HEADS, 1);
    // Layer 3: featB -> d_out[N,32], 1 head x 32 ch, no ELU (mean over 1 head = identity)
    run_layer(featB, wt3, 3, att3, b3, hbuf, outF, 1, OUT_DIM, 0);
}
